// DrugEncoder_61675730371301
// MI455X (gfx1250) — matrix operand order repaired and checked by reading, not compiled, not run on hardware
//
#include <hip/hip_runtime.h>
#include <hip/hip_bf16.h>
#include <stdint.h>

#define NN 100000
#define EE 1600000
#define GG 2048
#define HID 64

typedef __attribute__((ext_vector_type(16))) _Float16 v16h;
typedef __attribute__((ext_vector_type(8)))  float    v8f;

union Frag16 { v16h v; uint4 u[2]; };

__device__ __forceinline__ void atomicMaxF(float* addr, float val) {
    if (val >= 0.0f) atomicMax((int*)addr, __float_as_int(val));
    else             atomicMin((unsigned int*)addr, __float_as_uint(val));
}

// ---------------- utility kernels ----------------
__global__ void k_fill_f32(float* p, float v, int n) {
    int i = blockIdx.x * blockDim.x + threadIdx.x;
    if (i < n) p[i] = v;
}
__global__ void k_fill_u32(unsigned int* p, unsigned int v, int n) {
    int i = blockIdx.x * blockDim.x + threadIdx.x;
    if (i < n) p[i] = v;
}
__global__ void k_f32_to_f16(const float* __restrict__ in, _Float16* __restrict__ out, int n) {
    int i = blockIdx.x * blockDim.x + threadIdx.x;
    if (i < n) out[i] = (_Float16)in[i];
}

// Pack W (K x 64, f32 row-major) into WMMA B-fragment order:
// Bp[((kc*4 + tn)*32 + lane)*16 + e] = W[(kc*32 + 16*(lane/16) + e)*64 + tn*16 + lane%16]
__global__ void k_pack_b(const float* __restrict__ W, _Float16* __restrict__ Bp, int K) {
    int idx = blockIdx.x * blockDim.x + threadIdx.x;
    int total = (K >> 5) * 4 * 512;
    if (idx >= total) return;
    int e    = idx & 15;
    int lane = (idx >> 4) & 31;
    int tn   = (idx >> 9) & 3;
    int kc   = idx >> 11;
    int g = lane >> 4, n = lane & 15;
    int Kg  = kc * 32 + 16 * g + e;
    int col = tn * 16 + n;
    Bp[idx] = (_Float16)W[Kg * HID + col];
}

// wv = We @ att_e  (16 floats); collapses the edge-feature GEMM to a per-edge dot.
__global__ void k_wevec(const float* __restrict__ We, const float* __restrict__ atte,
                        float* __restrict__ wv) {
    int i = threadIdx.x;
    if (i < 16) {
        float s = 0.f;
        for (int j = 0; j < HID; ++j) s += We[i * HID + j] * atte[j];
        wv[i] = s;
    }
}

// ---------------- WMMA GEMM: C[M x 64] = A[M x K](f16) @ Bpack ----------------
// One wave owns a full 16x64 output strip: A fragment loaded ONCE per K-chunk,
// 4 accumulators, 4 WMMAs per A load (quarters A-side HBM traffic).
// K % 32 == 0, M % 16 == 0.  flags: bit0 = relu, bit1 = add bias
__global__ void __launch_bounds__(256)
k_wmma_gemm64(const _Float16* __restrict__ A, const _Float16* __restrict__ Bp,
              const float* __restrict__ bias, float* __restrict__ C,
              int M, int K, int flags) {
    int wave = (blockIdx.x * blockDim.x + threadIdx.x) >> 5;
    int lane = threadIdx.x & 31;
    int tilesM = M >> 4;
    if (wave >= tilesM) return;              // wave-uniform: EXEC stays all-1s for WMMA
    int g = lane >> 4, n = lane & 15;
    int row = (wave << 4) + n;               // A-fragment row = lane % 16

    v8f acc[4] = {{}, {}, {}, {}};
    int nkc = K >> 5;
    for (int kc = 0; kc < nkc; ++kc) {
        Frag16 a;
        const _Float16* abase = A + (size_t)row * K + kc * 32 + (g << 4);
        a.u[0] = *(const uint4*)(abase);          // K = kc*32 + 16g + [0..7]
        a.u[1] = *(const uint4*)(abase + 8);      // K = kc*32 + 16g + 8 + [0..7]
        const _Float16* bbase = Bp + ((size_t)kc * 2048) + ((size_t)lane << 4);
        Frag16 b0, b1, b2, b3;                    // tn stride = 512 halves
        b0.u[0] = *(const uint4*)(bbase);          b0.u[1] = *(const uint4*)(bbase + 8);
        b1.u[0] = *(const uint4*)(bbase + 512);    b1.u[1] = *(const uint4*)(bbase + 520);
        b2.u[0] = *(const uint4*)(bbase + 1024);   b2.u[1] = *(const uint4*)(bbase + 1032);
        b3.u[0] = *(const uint4*)(bbase + 1536);   b3.u[1] = *(const uint4*)(bbase + 1544);
        acc[0] = __builtin_amdgcn_wmma_f32_16x16x32_f16(false, a.v, false, b0.v, (short)0, acc[0], false, false);
        acc[1] = __builtin_amdgcn_wmma_f32_16x16x32_f16(false, a.v, false, b1.v, (short)0, acc[1], false, false);
        acc[2] = __builtin_amdgcn_wmma_f32_16x16x32_f16(false, a.v, false, b2.v, (short)0, acc[2], false, false);
        acc[3] = __builtin_amdgcn_wmma_f32_16x16x32_f16(false, a.v, false, b3.v, (short)0, acc[3], false, false);
    }
#pragma unroll
    for (int tn = 0; tn < 4; ++tn) {
        int col = (tn << 4) + n;
        float bv = (flags & 2) ? bias[col] : 0.0f;
#pragma unroll
        for (int r = 0; r < 8; ++r) {
            int m = (wave << 4) + r + (g << 3);  // C/D layout: M = r + 8*(lane/16)
            float v = acc[tn][r] + bv;
            if (flags & 1) v = fmaxf(v, 0.0f);
            C[(size_t)m * HID + col] = v;
        }
    }
}

// ---------------- graph kernels ----------------
__global__ void k_deg(const int* __restrict__ dst, float* __restrict__ deg, int E) {
    int e = blockIdx.x * blockDim.x + threadIdx.x;
    if (e < E) atomicAdd(&deg[dst[e]], 1.0f);
}
__global__ void k_rsqrt(float* __restrict__ d, int n) {
    int i = blockIdx.x * blockDim.x + threadIdx.x;
    if (i < n) d[i] = rsqrtf(d[i]);
}
__global__ void k_cnt(const int* __restrict__ dst, float* __restrict__ cnt, int E) {
    int e = blockIdx.x * blockDim.x + threadIdx.x;
    if (e < E) atomicAdd(&cnt[dst[e]], 1.0f);
}

// out[dst] += h[src] * dinv[src]*dinv[dst]  (16 threads/edge, float4 each)
__global__ void k_gcn_scatter(const float* __restrict__ h, const int* __restrict__ src,
                              const int* __restrict__ dst, const float* __restrict__ dinv,
                              float* __restrict__ out, int E) {
    int t = blockIdx.x * blockDim.x + threadIdx.x;
    int e = t >> 4;
    if (e >= E) return;
    int d0 = (t & 15) << 2;
    int s = src[e], d = dst[e];
    float nrm = dinv[s] * dinv[d];
    float4 hv = *(const float4*)(h + (size_t)s * HID + d0);
    float* o = out + (size_t)d * HID + d0;
    atomicAdd(o + 0, hv.x * nrm);
    atomicAdd(o + 1, hv.y * nrm);
    atomicAdd(o + 2, hv.z * nrm);
    atomicAdd(o + 3, hv.w * nrm);
}

__global__ void k_gcn_self_finalize(const float* __restrict__ h, const float* __restrict__ dinv,
                                    const float* __restrict__ bias, float* __restrict__ out,
                                    _Float16* __restrict__ out16, int n) {
    int t = blockIdx.x * blockDim.x + threadIdx.x;
    int i = t >> 6;
    if (i >= n) return;
    int d = t & 63;
    float v = out[(size_t)i * HID + d] + h[(size_t)i * HID + d] * dinv[i] * dinv[i] + bias[d];
    v = fmaxf(v, 0.0f);
    out[(size_t)i * HID + d] = v;
    out16[(size_t)i * HID + d] = (_Float16)v;
}

__global__ void k_attn_node(const float* __restrict__ hW, const float* __restrict__ atts,
                            const float* __restrict__ attd, float* __restrict__ as_,
                            float* __restrict__ ad_, int n) {
    int i = blockIdx.x * blockDim.x + threadIdx.x;
    if (i >= n) return;
    float s = 0.f, d = 0.f;
    for (int j = 0; j < HID; ++j) {
        float v = hW[(size_t)i * HID + j];
        s += v * atts[j];
        d += v * attd[j];
    }
    as_[i] = s;
    ad_[i] = d;
}

__global__ void k_ae_edge(const float* __restrict__ ea, const float* __restrict__ wv,
                          const int* __restrict__ dst, float* __restrict__ ae,
                          float* __restrict__ ae_sum, int E) {
    int e = blockIdx.x * blockDim.x + threadIdx.x;
    if (e >= E) return;
    float s = 0.f;
    for (int j = 0; j < 16; ++j) s += ea[(size_t)e * 16 + j] * wv[j];
    ae[e] = s;
    atomicAdd(&ae_sum[dst[e]], s);
}
__global__ void k_ae_self_div(float* __restrict__ ae_sum, const float* __restrict__ cnt, int n) {
    int i = blockIdx.x * blockDim.x + threadIdx.x;
    if (i < n) ae_sum[i] = ae_sum[i] / fmaxf(cnt[i], 1.0f);
}

__global__ void k_att_logits_edge(const float* __restrict__ as_, const float* __restrict__ ad_,
                                  const float* __restrict__ ae, const int* __restrict__ src,
                                  const int* __restrict__ dst, float* __restrict__ aval,
                                  float* __restrict__ amax, int E) {
    int e = blockIdx.x * blockDim.x + threadIdx.x;
    if (e >= E) return;
    float a = as_[src[e]] + ad_[dst[e]] + ae[e];
    a = a > 0.f ? a : 0.2f * a;
    aval[e] = a;
    atomicMaxF(&amax[dst[e]], a);
}
__global__ void k_att_logits_self(const float* __restrict__ as_, const float* __restrict__ ad_,
                                  const float* __restrict__ ae_self, float* __restrict__ aval,
                                  float* __restrict__ amax, int n) {
    int i = blockIdx.x * blockDim.x + threadIdx.x;
    if (i >= n) return;
    float a = as_[i] + ad_[i] + ae_self[i];
    a = a > 0.f ? a : 0.2f * a;
    aval[EE + i] = a;
    atomicMaxF(&amax[i], a);
}

__global__ void k_exp_edge(float* __restrict__ aval, const int* __restrict__ dst,
                           const float* __restrict__ amax, float* __restrict__ den, int E) {
    int e = blockIdx.x * blockDim.x + threadIdx.x;
    if (e >= E) return;
    int d = dst[e];
    float ex = __expf(aval[e] - amax[d]);
    aval[e] = ex;
    atomicAdd(&den[d], ex);
}
__global__ void k_exp_self(float* __restrict__ aval, const float* __restrict__ amax,
                           float* __restrict__ den, int n) {
    int i = blockIdx.x * blockDim.x + threadIdx.x;
    if (i >= n) return;
    float ex = __expf(aval[EE + i] - amax[i]);
    aval[EE + i] = ex;
    atomicAdd(&den[i], ex);
}

__global__ void k_gat_scatter(const float* __restrict__ hW, const int* __restrict__ src,
                              const int* __restrict__ dst, const float* __restrict__ aval,
                              const float* __restrict__ den, float* __restrict__ out, int E) {
    int t = blockIdx.x * blockDim.x + threadIdx.x;
    int e = t >> 4;
    if (e >= E) return;
    int d0 = (t & 15) << 2;
    int s = src[e], d = dst[e];
    float alpha = aval[e] / den[d];
    float4 hv = *(const float4*)(hW + (size_t)s * HID + d0);
    float* o = out + (size_t)d * HID + d0;
    atomicAdd(o + 0, hv.x * alpha);
    atomicAdd(o + 1, hv.y * alpha);
    atomicAdd(o + 2, hv.z * alpha);
    atomicAdd(o + 3, hv.w * alpha);
}

__global__ void k_gat_self_finalize(const float* __restrict__ hW, const float* __restrict__ aval,
                                    const float* __restrict__ den, const float* __restrict__ bias,
                                    float* __restrict__ out, _Float16* __restrict__ out16, int n) {
    int t = blockIdx.x * blockDim.x + threadIdx.x;
    int i = t >> 6;
    if (i >= n) return;
    int d = t & 63;
    float alpha = aval[EE + i] / den[i];
    float v = out[(size_t)i * HID + d] + hW[(size_t)i * HID + d] * alpha + bias[d];
    v = fmaxf(v, 0.0f);
    out[(size_t)i * HID + d] = v;
    out16[(size_t)i * HID + d] = (_Float16)v;
}

__global__ void k_pool(const float* __restrict__ h, const int* __restrict__ batch,
                       float* __restrict__ g, int n) {
    int t = blockIdx.x * blockDim.x + threadIdx.x;
    int i = t >> 6;
    if (i >= n) return;
    int d = t & 63;
    atomicMaxF(&g[(size_t)batch[i] * HID + d], h[(size_t)i * HID + d]);
}

// ---------------- host side ----------------
static inline int gb(long n, int t) { return (int)((n + t - 1) / t); }

struct GatWs {
    const int *src, *dst;
    const float *ea;
    float *hA, *hB, *as_, *ad_, *ae_edge, *ae_sum, *aval, *amax, *den, *cnt;
    _Float16 *h16;
};

static void run_gat(const GatWs& w, const _Float16* Wpack,
                    const float* wv, const float* atts, const float* attd, const float* bias,
                    hipStream_t stream) {
    // hA = h16 @ W  (WMMA, one wave per 16x64 strip)
    k_wmma_gemm64<<<gb((long)(NN / 16) * 32, 256), 256, 0, stream>>>(w.h16, Wpack, nullptr, w.hA, NN, HID, 0);
    k_attn_node<<<gb(NN, 256), 256, 0, stream>>>(w.hA, atts, attd, w.as_, w.ad_, NN);
    k_fill_f32<<<gb(NN, 256), 256, 0, stream>>>(w.ae_sum, 0.f, NN);
    k_ae_edge<<<gb(EE, 256), 256, 0, stream>>>(w.ea, wv, w.dst, w.ae_edge, w.ae_sum, EE);
    k_ae_self_div<<<gb(NN, 256), 256, 0, stream>>>(w.ae_sum, w.cnt, NN);
    k_fill_u32<<<gb(NN, 256), 256, 0, stream>>>((unsigned int*)w.amax, 0xFF7FFFFFu, NN);
    k_att_logits_edge<<<gb(EE, 256), 256, 0, stream>>>(w.as_, w.ad_, w.ae_edge, w.src, w.dst, w.aval, w.amax, EE);
    k_att_logits_self<<<gb(NN, 256), 256, 0, stream>>>(w.as_, w.ad_, w.ae_sum, w.aval, w.amax, NN);
    k_fill_f32<<<gb(NN, 256), 256, 0, stream>>>(w.den, 0.f, NN);
    k_exp_edge<<<gb(EE, 256), 256, 0, stream>>>(w.aval, w.dst, w.amax, w.den, EE);
    k_exp_self<<<gb(NN, 256), 256, 0, stream>>>(w.aval, w.amax, w.den, NN);
    k_fill_f32<<<gb((long)NN * HID, 256), 256, 0, stream>>>(w.hB, 0.f, NN * HID);
    k_gat_scatter<<<gb((long)EE * 16, 256), 256, 0, stream>>>(w.hA, w.src, w.dst, w.aval, w.den, w.hB, EE);
    k_gat_self_finalize<<<gb((long)NN * HID, 256), 256, 0, stream>>>(w.hA, w.aval, w.den, bias, w.hB, w.h16, NN);
}

extern "C" void kernel_launch(void* const* d_in, const int* in_sizes, int n_in,
                              void* d_out, int out_size, void* d_ws, size_t ws_size,
                              hipStream_t stream) {
    const float* x     = (const float*)d_in[0];
    const int*   ei    = (const int*)d_in[1];
    const int*   src   = ei;
    const int*   dst   = ei + EE;
    const float* ea    = (const float*)d_in[2];
    const int*   batch = (const int*)d_in[3];
    const float* W_gcn = (const float*)d_in[4];
    const float* b_gcn = (const float*)d_in[5];
    const float* W_g1  = (const float*)d_in[6];
    const float* We_g1 = (const float*)d_in[7];
    const float* as_g1 = (const float*)d_in[8];
    const float* ad_g1 = (const float*)d_in[9];
    const float* ae_g1 = (const float*)d_in[10];
    const float* b_g1  = (const float*)d_in[11];
    const float* W_g2  = (const float*)d_in[12];
    const float* We_g2 = (const float*)d_in[13];
    const float* as_g2 = (const float*)d_in[14];
    const float* ad_g2 = (const float*)d_in[15];
    const float* ae_g2 = (const float*)d_in[16];
    const float* b_g2  = (const float*)d_in[17];
    const float* W_fc1 = (const float*)d_in[18];
    const float* b_fc1 = (const float*)d_in[19];
    const float* W_fc2 = (const float*)d_in[20];
    const float* b_fc2 = (const float*)d_in[21];

    // carve workspace
    char* base = (char*)d_ws;
    size_t off = 0;
    auto alloc = [&](size_t bytes) -> void* {
        off = (off + 255) & ~(size_t)255;
        void* p = base + off;
        off += bytes;
        return p;
    };
    _Float16* xh      = (_Float16*)alloc((size_t)NN * 128 * 2);
    _Float16* h16     = (_Float16*)alloc((size_t)NN * HID * 2);
    float*    hA      = (float*)alloc((size_t)NN * HID * 4);
    float*    hB      = (float*)alloc((size_t)NN * HID * 4);
    float*    dinv    = (float*)alloc((size_t)NN * 4);
    float*    cnt     = (float*)alloc((size_t)NN * 4);
    float*    as_     = (float*)alloc((size_t)NN * 4);
    float*    ad_     = (float*)alloc((size_t)NN * 4);
    float*    ae_edge = (float*)alloc((size_t)EE * 4);
    float*    ae_sum  = (float*)alloc((size_t)NN * 4);
    float*    aval    = (float*)alloc((size_t)(EE + NN) * 4);
    float*    amax    = (float*)alloc((size_t)NN * 4);
    float*    den     = (float*)alloc((size_t)NN * 4);
    _Float16* p_gcn   = (_Float16*)alloc(4 * 4 * 512 * 2);
    _Float16* p_g1    = (_Float16*)alloc(2 * 4 * 512 * 2);
    _Float16* p_g2    = (_Float16*)alloc(2 * 4 * 512 * 2);
    _Float16* p_f1    = (_Float16*)alloc(2 * 4 * 512 * 2);
    _Float16* p_f2    = (_Float16*)alloc(2 * 4 * 512 * 2);
    float*    wv1     = (float*)alloc(16 * 4);
    float*    wv2     = (float*)alloc(16 * 4);
    float*    g       = (float*)alloc((size_t)GG * HID * 4);
    _Float16* g16     = (_Float16*)alloc((size_t)GG * HID * 2);
    float*    fc1o    = (float*)alloc((size_t)GG * HID * 4);
    _Float16* fc1h    = (_Float16*)alloc((size_t)GG * HID * 2);

    // --- weight packing + precompute ---
    k_pack_b<<<gb(4 * 4 * 512, 256), 256, 0, stream>>>(W_gcn, p_gcn, 128);
    k_pack_b<<<gb(2 * 4 * 512, 256), 256, 0, stream>>>(W_g1, p_g1, 64);
    k_pack_b<<<gb(2 * 4 * 512, 256), 256, 0, stream>>>(W_g2, p_g2, 64);
    k_pack_b<<<gb(2 * 4 * 512, 256), 256, 0, stream>>>(W_fc1, p_f1, 64);
    k_pack_b<<<gb(2 * 4 * 512, 256), 256, 0, stream>>>(W_fc2, p_f2, 64);
    k_wevec<<<1, 64, 0, stream>>>(We_g1, ae_g1, wv1);
    k_wevec<<<1, 64, 0, stream>>>(We_g2, ae_g2, wv2);
    k_f32_to_f16<<<gb((long)NN * 128, 256), 256, 0, stream>>>(x, xh, NN * 128);

    // --- degrees (with self loop) and GAT in-degree counts ---
    k_fill_f32<<<gb(NN, 256), 256, 0, stream>>>(dinv, 1.0f, NN);
    k_deg<<<gb(EE, 256), 256, 0, stream>>>(dst, dinv, EE);
    k_rsqrt<<<gb(NN, 256), 256, 0, stream>>>(dinv, NN);
    k_fill_f32<<<gb(NN, 256), 256, 0, stream>>>(cnt, 0.f, NN);
    k_cnt<<<gb(EE, 256), 256, 0, stream>>>(dst, cnt, EE);

    // --- GCN: hA = x @ W_gcn (WMMA) ; sym-norm aggregate ; +b ; relu ---
    k_wmma_gemm64<<<gb((long)(NN / 16) * 32, 256), 256, 0, stream>>>(xh, p_gcn, nullptr, hA, NN, 128, 0);
    k_fill_f32<<<gb((long)NN * HID, 256), 256, 0, stream>>>(hB, 0.f, NN * HID);
    k_gcn_scatter<<<gb((long)EE * 16, 256), 256, 0, stream>>>(hA, src, dst, dinv, hB, EE);
    k_gcn_self_finalize<<<gb((long)NN * HID, 256), 256, 0, stream>>>(hA, dinv, b_gcn, hB, h16, NN);

    // --- GAT layers ---
    GatWs w{src, dst, ea, hA, hB, as_, ad_, ae_edge, ae_sum, aval, amax, den, cnt, h16};
    run_gat(w, p_g1, wv1, as_g1, ad_g1, b_g1, stream);
    run_gat(w, p_g2, wv2, as_g2, ad_g2, b_g2, stream);

    // --- global max pool per graph ---
    k_fill_u32<<<gb((long)GG * HID, 256), 256, 0, stream>>>((unsigned int*)g, 0xFF7FFFFFu, GG * HID);
    k_pool<<<gb((long)NN * HID, 256), 256, 0, stream>>>(hB, batch, g, NN);
    k_f32_to_f16<<<gb((long)GG * HID, 256), 256, 0, stream>>>(g, g16, GG * HID);

    // --- FC head (WMMA): relu(g@W1+b1) @ W2 + b2 ---
    k_wmma_gemm64<<<gb((long)(GG / 16) * 32, 256), 256, 0, stream>>>(g16, p_f1, b_fc1, fc1o, GG, HID, 3);
    k_f32_to_f16<<<gb((long)GG * HID, 256), 256, 0, stream>>>(fc1o, fc1h, GG * HID);
    k_wmma_gemm64<<<gb((long)(GG / 16) * 32, 256), 256, 0, stream>>>(fc1h, p_f2, b_fc2, (float*)d_out, GG, HID, 2);
}
